// DVGGA_67551245631659
// MI455X (gfx1250) — matrix operand order repaired and checked by reading, not compiled, not run on hardware
//
#include <hip/hip_runtime.h>
#include <hip/hip_bf16.h>
#include <math.h>

// ---------------------------------------------------------------------------
// Problem constants (match reference)
// ---------------------------------------------------------------------------
constexpr int G  = 512;     // graphs
constexpr int NN = 512;     // nodes per graph
constexpr int EE = 2048;    // edges per graph
constexpr int FF = 64;      // node features
constexpr int D1 = 128;     // first gcn / dense dim
constexpr int D2 = 64;      // second gcn dim
constexpr int LL = 32;      // labels
constexpr int PP = 16384;   // pos edges among graph embeddings

// ---------------------------------------------------------------------------
// WMMA helpers (CDNA5 gfx1250, wave32). v_wmma_f32_16x16x32_f16:
//   D[16x16,f32] = A[16x32,f16] x B[32x16,f16] + C
// Per-lane layouts per CDNA5 ISA 7.12.2.
// ---------------------------------------------------------------------------
typedef __attribute__((ext_vector_type(16))) _Float16 v16h;
typedef __attribute__((ext_vector_type(8)))  _Float16 v8h;
typedef __attribute__((ext_vector_type(8)))  float    v8f;

// A fragment: row-major f16 matrix `A` (leading dim lda halfs), 16 rows,
// K window [k0,k0+32). Lane l: m = l%16, hi = l/16.
// Per-lane elements are two contiguous 8-half (16 B) runs:
//   elems 0..7  <-> K = k0 + 8*hi + 0..7
//   elems 8..15 <-> K = k0 + 16 + 8*hi + 0..7
// Loaded as two b128s (all our lda values keep these 16 B aligned).
__device__ inline v16h frag_a_f16(const _Float16* __restrict__ A, int lda, int k0) {
  const int l  = threadIdx.x & 31;
  const int m  = l & 15;
  const int hi = l >> 4;
  const _Float16* base = A + m * lda + k0 + 8 * hi;
  const v8h lo = *(const v8h*)(base);
  const v8h hv = *(const v8h*)(base + 16);
  v16h a;
#pragma unroll
  for (int i = 0; i < 8; ++i) { a[i] = lo[i]; a[i + 8] = hv[i]; }
  return a;
}

// B fragment from row-major f32 weight matrix W[K x ldb], cols [n0,n0+16),
// K window [k0,k0+32). Lane l: n = n0 + l%16, hi = l/16; elems 0..7 <-> K 8*hi+0..7,
// elems 8..15 <-> K 16+8*hi+0..7 (relative to k0). Converted to f16 on the fly.
// (Weights are tiny and L2-hot; lanes 0-15 are coalesced per K row.)
__device__ inline v16h frag_b_f32(const float* __restrict__ W, int ldb, int k0, int n0) {
  const int l  = threadIdx.x & 31;
  const int n  = n0 + (l & 15);
  const int hi = l >> 4;
  v16h b;
#pragma unroll
  for (int i = 0; i < 16; ++i) {
    const int k = k0 + 8 * hi + (i & 7) + 16 * (i >> 3);
    b[i] = (_Float16)W[k * ldb + n];
  }
  return b;
}

#define WMMA_F32_F16(a, b, c) \
  __builtin_amdgcn_wmma_f32_16x16x32_f16(false, (a), false, (b), (short)0, (c), false, false)

// ---------------------------------------------------------------------------
// Kernel 1: per-graph aggregation.
//   deg[n]  = 1 + #(dst==n);  dinv = rsqrt(deg)
//   c[n]    = dinv[n] * sum_{e:src=n} dinv[dst_e] + dinv[n]^2   (self loop)
//   y[g,f]  = (1/16) * sum_n c[n] * x[g,n,f]            (stored as f16)
// One block per graph; degrees / coefficients via LDS atomics (ds_add).
// Feature stream: 16 row-partitions x 16 float4 lanes -> each wave issues
// global_load_b128 over 512 contiguous bytes (feeds the 23.3 TB/s HBM path).
// ---------------------------------------------------------------------------
__global__ void k1_graph_aggregate(const float* __restrict__ feat,
                                   const int*   __restrict__ edges,
                                   _Float16*    __restrict__ yh) {
  __shared__ float s_dinv[NN];
  __shared__ float s_c[NN];
  __shared__ int   s_deg[NN];
  __shared__ float s_red[16][FF];              // 16 partitions x 64 feats

  const int g   = blockIdx.x;
  const int tid = threadIdx.x;                 // 256 threads = 8 waves
  const int* src = edges + (size_t)g * 2 * EE;
  const int* dst = src + EE;

  for (int n = tid; n < NN; n += 256) s_deg[n] = 1;   // self loop
  __syncthreads();
  for (int e = tid; e < EE; e += 256) atomicAdd(&s_deg[dst[e]], 1);
  __syncthreads();
  for (int n = tid; n < NN; n += 256) {
    s_dinv[n] = rsqrtf((float)s_deg[n]);
    s_c[n]    = 0.0f;
  }
  __syncthreads();
  for (int e = tid; e < EE; e += 256) atomicAdd(&s_c[src[e]], s_dinv[dst[e]]);
  __syncthreads();
  for (int n = tid; n < NN; n += 256) {
    const float di = s_dinv[n];
    s_c[n] = di * s_c[n] + di * di;            // edge term + self-loop term
  }
  __syncthreads();

  // y[f] = sum_n c[n] * x[n,f]
  const int lq   = tid & 15;                   // float4 lane: feats lq*4..+3
  const int part = tid >> 4;                   // row partition 0..15
  const float4* xg4 = (const float4*)(feat + (size_t)g * NN * FF);
  float4 acc = make_float4(0.f, 0.f, 0.f, 0.f);
  for (int n = part; n < NN; n += 16) {
    const float4 x = xg4[n * (FF / 4) + lq];   // b128, contiguous per wave
    const float  cn = s_c[n];
    acc.x = fmaf(cn, x.x, acc.x);
    acc.y = fmaf(cn, x.y, acc.y);
    acc.z = fmaf(cn, x.z, acc.z);
    acc.w = fmaf(cn, x.w, acc.w);
  }
  s_red[part][lq * 4 + 0] = acc.x;
  s_red[part][lq * 4 + 1] = acc.y;
  s_red[part][lq * 4 + 2] = acc.z;
  s_red[part][lq * 4 + 3] = acc.w;
  __syncthreads();
  if (tid < FF) {
    float s = 0.0f;
#pragma unroll
    for (int p = 0; p < 16; ++p) s += s_red[p][tid];
    yh[g * FF + tid] = (_Float16)(s * 0.0625f);  // fold 1/16 (mean over K)
  }
}

// ---------------------------------------------------------------------------
// Kernel 2 (fused WMMA x2):
//   emb[16,128] = y[16,64] @ W1[64,128] + 32*b1     (kept in LDS as f16)
//   t1 [16,128] = emb @ conv1_W[128,128]            (f32 to workspace)
// One block = 16 output rows, 8 waves, one 16-col tile per wave.
// ---------------------------------------------------------------------------
__global__ void k2_emb_conv1(const _Float16* __restrict__ yh,
                             const float*    __restrict__ W1,
                             const float*    __restrict__ b1,
                             const float*    __restrict__ conv1W,
                             float*          __restrict__ t1) {
  __shared__ _Float16 s_emb[16][D1 + 8];       // lda=136 (16 B-aligned rows)
  const int m0 = blockIdx.x * 16;
  const int w  = threadIdx.x >> 5;             // wave 0..7 -> 16-col tile
  const int n0 = w * 16;
  const int l  = threadIdx.x & 31;
  const int lm = l & 15, lh = l >> 4;

  // GEMM 1: K = 64 -> 2 wmma steps
  v8f c = {};
#pragma unroll
  for (int k0 = 0; k0 < FF; k0 += 32) {
    const v16h a = frag_a_f16(yh + (size_t)m0 * FF, FF, k0);
    const v16h b = frag_b_f32(W1, D1, k0, n0);
    c = WMMA_F32_F16(a, b, c);
  }
  {
    const int   n    = n0 + lm;
    const float bias = 32.0f * b1[n];          // N/K * b1 from the collapse
#pragma unroll
    for (int r = 0; r < 8; ++r)
      s_emb[lh * 8 + r][n] = (_Float16)(c[r] + bias);
  }
  __syncthreads();

  // GEMM 2: K = 128 -> 4 wmma steps, A from LDS f16 tile
  v8f c2 = {};
#pragma unroll
  for (int k0 = 0; k0 < D1; k0 += 32) {
    const v16h a = frag_a_f16(&s_emb[0][0], D1 + 8, k0);
    const v16h b = frag_b_f32(conv1W, D1, k0, n0);
    c2 = WMMA_F32_F16(a, b, c2);
  }
  const int n = n0 + lm;
#pragma unroll
  for (int r = 0; r < 8; ++r)
    t1[(size_t)(m0 + lh * 8 + r) * D1 + n] = c2[r];
}

// ---------------------------------------------------------------------------
// Kernel 3a: degrees of the pos_edges graph (512 nodes, 16384 edges).
// ---------------------------------------------------------------------------
__global__ void k3a_deg2(const int* __restrict__ pos, float* __restrict__ dinv2) {
  __shared__ int s_deg[G];
  const int tid = threadIdx.x;                 // 512 threads, 1 block
  s_deg[tid] = 1;                              // self loop
  __syncthreads();
  const int* d = pos + PP;
  for (int e = tid; e < PP; e += 512) atomicAdd(&s_deg[d[e]], 1);
  __syncthreads();
  dinv2[tid] = rsqrtf((float)s_deg[tid]);
}

// ---------------------------------------------------------------------------
// Kernel 3/5: GCN propagation over pos_edges, LDS-resident accumulator.
//   out[d,:] = sum_e dinv[s]*dinv[d]*x[s,:] + dinv[d]^2*x[d,:] + bias ; relu?
// One block per 32-column tile. 32 edge-groups x 8 float4 column-lanes:
// row gathers are global_load_b128; scatter-add stays in LDS (ds_add_f32).
// Output written as f16 (A operand of the next WMMA GEMM).
// ---------------------------------------------------------------------------
__global__ void k_propagate(const float* __restrict__ xin, int ncols,
                            const int*   __restrict__ pos,
                            const float* __restrict__ dinv2,
                            const float* __restrict__ bias,
                            _Float16*    __restrict__ outh, int do_relu) {
  __shared__ float s_acc[G][32];               // 64 KB
  const int c0  = blockIdx.x * 32;
  const int tid = threadIdx.x;                 // 256 threads
  const int j4  = tid & 7;                     // float4 lane: cols j4*4..+3
  const int eg  = tid >> 3;                    // edge group 0..31
  const int nc4 = ncols >> 2;
  const int jj  = j4 * 4;
  const float4* x4 = (const float4*)xin;
  const int cb4 = (c0 >> 2) + j4;              // float4 index within a row

  // init with self-loop term
  for (int n = eg; n < G; n += 32) {
    const float  di = dinv2[n];
    const float4 v  = x4[n * nc4 + cb4];
    const float  d2 = di * di;
    s_acc[n][jj + 0] = d2 * v.x;
    s_acc[n][jj + 1] = d2 * v.y;
    s_acc[n][jj + 2] = d2 * v.z;
    s_acc[n][jj + 3] = d2 * v.w;
  }
  __syncthreads();

  const int* ps = pos;
  const int* pd = pos + PP;
  for (int e = eg; e < PP; e += 32) {
    const int    s   = ps[e], d = pd[e];
    const float  wgt = dinv2[s] * dinv2[d];
    const float4 v   = x4[s * nc4 + cb4];      // b128 gather
    atomicAdd(&s_acc[d][jj + 0], wgt * v.x);
    atomicAdd(&s_acc[d][jj + 1], wgt * v.y);
    atomicAdd(&s_acc[d][jj + 2], wgt * v.z);
    atomicAdd(&s_acc[d][jj + 3], wgt * v.w);
  }
  __syncthreads();

  for (int n = eg; n < G; n += 32) {
#pragma unroll
    for (int k = 0; k < 4; ++k) {
      float v = s_acc[n][jj + k] + bias[c0 + jj + k];
      if (do_relu) v = fmaxf(v, 0.0f);
      outh[(size_t)n * ncols + c0 + jj + k] = (_Float16)v;
    }
  }
}

// ---------------------------------------------------------------------------
// Kernel 4: t2[512,64] = h2[512,128] @ mu_W[128,64]   (WMMA, f32 out)
// ---------------------------------------------------------------------------
__global__ void k4_mu_gemm(const _Float16* __restrict__ h2h,
                           const float*    __restrict__ muW,
                           float*          __restrict__ t2) {
  const int m0 = blockIdx.x * 16;
  const int w  = threadIdx.x >> 5;             // 4 waves -> 4 x 16 cols
  const int n0 = w * 16;
  const int l  = threadIdx.x & 31;
  const int lm = l & 15, lh = l >> 4;
  v8f c = {};
#pragma unroll
  for (int k0 = 0; k0 < D1; k0 += 32) {
    const v16h a = frag_a_f16(h2h + (size_t)m0 * D1, D1, k0);
    const v16h b = frag_b_f32(muW, D2, k0, n0);
    c = WMMA_F32_F16(a, b, c);
  }
  const int n = n0 + lm;
#pragma unroll
  for (int r = 0; r < 8; ++r)
    t2[(size_t)(m0 + lh * 8 + r) * D2 + n] = c[r];
}

// ---------------------------------------------------------------------------
// Kernel 6: predictions = log_softmax(mu @ clf_W + clf_b)   [512 x 32]
// WMMA GEMM into LDS, then row-wise log-sum-exp.
// ---------------------------------------------------------------------------
__global__ void k6_classify(const _Float16* __restrict__ muh,
                            const float*    __restrict__ clfW,
                            const float*    __restrict__ clfb,
                            float*          __restrict__ out) {
  __shared__ float s_log[16][LL];
  const int m0 = blockIdx.x * 16;
  const int w  = threadIdx.x >> 5;             // 2 waves -> 2 x 16 cols
  const int n0 = w * 16;
  const int l  = threadIdx.x & 31;
  const int lm = l & 15, lh = l >> 4;
  v8f c = {};
#pragma unroll
  for (int k0 = 0; k0 < D2; k0 += 32) {
    const v16h a = frag_a_f16(muh + (size_t)m0 * D2, D2, k0);
    const v16h b = frag_b_f32(clfW, LL, k0, n0);
    c = WMMA_F32_F16(a, b, c);
  }
  const int n = n0 + lm;
#pragma unroll
  for (int r = 0; r < 8; ++r)
    s_log[lh * 8 + r][n] = c[r] + clfb[n];
  __syncthreads();

  if (threadIdx.x < 16) {
    const int r = threadIdx.x;
    float mx = -1e30f;
#pragma unroll
    for (int jj = 0; jj < LL; ++jj) mx = fmaxf(mx, s_log[r][jj]);
    float sum = 0.0f;
#pragma unroll
    for (int jj = 0; jj < LL; ++jj) sum += __expf(s_log[r][jj] - mx);
    const float lse = mx + __logf(sum);
#pragma unroll
    for (int jj = 0; jj < LL; ++jj)
      out[(size_t)(m0 + r) * LL + jj] = s_log[r][jj] - lse;
  }
}

// ---------------------------------------------------------------------------
// Workspace layout (bytes, all 256-aligned). Total ~642 KB.
// ---------------------------------------------------------------------------
constexpr size_t OFF_YH    = 0;                      // f16 [512*64]   =  64 KB
constexpr size_t OFF_T1    = OFF_YH    + 65536;      // f32 [512*128]  = 256 KB
constexpr size_t OFF_DINV2 = OFF_T1    + 262144;     // f32 [512]      =   2 KB
constexpr size_t OFF_H2H   = OFF_DINV2 + 2048;       // f16 [512*128]  = 128 KB
constexpr size_t OFF_T2    = OFF_H2H   + 131072;     // f32 [512*64]   = 128 KB
constexpr size_t OFF_MUH   = OFF_T2    + 131072;     // f16 [512*64]   =  64 KB

extern "C" void kernel_launch(void* const* d_in, const int* in_sizes, int n_in,
                              void* d_out, int out_size, void* d_ws, size_t ws_size,
                              hipStream_t stream) {
  (void)in_sizes; (void)n_in; (void)out_size; (void)ws_size;

  const float* feat   = (const float*)d_in[0];   // [512,512,64]
  const int*   edges  = (const int*)  d_in[1];   // [512,2,2048]
  const int*   pos    = (const int*)  d_in[2];   // [2,16384]
  const float* W1     = (const float*)d_in[3];   // [64,128]
  const float* b1     = (const float*)d_in[4];   // [128]
  // d_in[5..8] = fc1_W, fc1_b, fc2_W, fc2_b — provably dead: softmax rows sum
  // to 1, so mean_k(assign^T h) == (1/16) sum_n h[n,:] independent of assign.
  const float* conv1W = (const float*)d_in[9];   // [128,128]
  const float* conv1b = (const float*)d_in[10];  // [128]
  const float* muW    = (const float*)d_in[11];  // [128,64]
  const float* mub    = (const float*)d_in[12];  // [64]
  const float* clfW   = (const float*)d_in[13];  // [64,32]
  const float* clfb   = (const float*)d_in[14];  // [32]
  float* out = (float*)d_out;                    // [512,32]

  char* ws = (char*)d_ws;
  _Float16* yh    = (_Float16*)(ws + OFF_YH);
  float*    t1    = (float*)   (ws + OFF_T1);
  float*    dinv2 = (float*)   (ws + OFF_DINV2);
  _Float16* h2h   = (_Float16*)(ws + OFF_H2H);
  float*    t2    = (float*)   (ws + OFF_T2);
  _Float16* muh   = (_Float16*)(ws + OFF_MUH);

  // 1. per-graph aggregation (streams the 64 MB feature tensor once, b128)
  k1_graph_aggregate<<<G, 256, 0, stream>>>(feat, edges, yh);
  // 2. emb = y@W1 + 32*b1 ; t1 = emb@conv1_W   (fused WMMA)
  k2_emb_conv1<<<G / 16, 256, 0, stream>>>(yh, W1, b1, conv1W, t1);
  // 3. pos-graph degrees, then h2 = relu(Ahat @ t1 + conv1_b)
  k3a_deg2<<<1, G, 0, stream>>>(pos, dinv2);
  k_propagate<<<D1 / 32, 256, 0, stream>>>(t1, D1, pos, dinv2, conv1b, h2h, 1);
  // 4. t2 = h2 @ mu_W
  k4_mu_gemm<<<G / 16, 128, 0, stream>>>(h2h, muW, t2);
  // 5. mu = Ahat @ t2 + mu_b
  k_propagate<<<D2 / 32, 256, 0, stream>>>(t2, D2, pos, dinv2, mub, muh, 0);
  // 6. predictions = log_softmax(mu @ clf_W + clf_b)
  k6_classify<<<G / 16, 64, 0, stream>>>(muh, clfW, clfb, out);
}
